// MoE_85315230368423
// MI455X (gfx1250) — matrix operand order repaired and checked by reading, not compiled, not run on hardware
//
#include <hip/hip_runtime.h>
#include <hip/hip_bf16.h>
#include <math.h>

// ---------------------------------------------------------------------------
// MoE (top-2 of 8 experts), B=4 S=2048 D=1024 F=4096, fp32 in/out.
// Strategy: route -> deterministic compaction -> bf16 WMMA expert GEMMs.
// ---------------------------------------------------------------------------

typedef __bf16 bf16_t;
typedef __attribute__((ext_vector_type(8)))  __bf16 bf16x8;
typedef __attribute__((ext_vector_type(16))) __bf16 v16bf;
typedef __attribute__((ext_vector_type(8)))  float  v8f;

#define NTOK   8192      // B*S
#define DDIM   1024
#define FDIM   4096
#define NEXP   8
#define TILE_M 16
#define HPAD   8                                  // 16B pad per LDS row
#define HSTRIDE (FDIM + HPAD)                     // 4104 elements
#define HBYTES  (TILE_M * HSTRIDE * 2)            // 131328 bytes dynamic LDS
#define INV_TEMP (1.0f / (1.0f + 1e-6f))

// ---- WMMA fragment helpers (CDNA5 16x16x32 bf16 layouts) -------------------
// A (16x32, MxK): lane r=lane&15 holds row r; elems 0..7 = K k0+half*8..+7,
//                 elems 8..15 = K k0+16+half*8..+7  (half = lane>>4)
static __device__ __forceinline__ v16bf load_a16x32(const bf16_t* row, int k0, int half) {
  bf16x8 lo = *(const bf16x8*)(row + k0 + half * 8);
  bf16x8 hi = *(const bf16x8*)(row + k0 + 16 + half * 8);
  return __builtin_shufflevector(lo, hi, 0,1,2,3,4,5,6,7,8,9,10,11,12,13,14,15);
}
// B (32x16, KxN) stored as B^T row-major [N][K]: lane n=lane&15 holds col n;
// elems 0..7 = K k0+half*8..+7, elems 8..15 = K k0+16+half*8..+7 (matches A)
static __device__ __forceinline__ v16bf load_b32x16(const bf16_t* rowN, int k0, int half) {
  bf16x8 lo = *(const bf16x8*)(rowN + k0 + half * 8);
  bf16x8 hi = *(const bf16x8*)(rowN + k0 + 16 + half * 8);
  return __builtin_shufflevector(lo, hi, 0,1,2,3,4,5,6,7,8,9,10,11,12,13,14,15);
}
static __device__ __forceinline__ v8f wmma_bf16(v16bf a, v16bf b, v8f c) {
  return __builtin_amdgcn_wmma_f32_16x16x32_bf16(false, a, false, b, (short)0, c, false, false);
}

// ---- small utility kernels -------------------------------------------------
__global__ void zero_kernel(float* p, int n) {
  int i = blockIdx.x * 256 + threadIdx.x;
  if (i < n) p[i] = 0.0f;
}
__global__ void cvt_bf16_kernel(const float* __restrict__ in, bf16_t* __restrict__ o, int n) {
  int i = blockIdx.x * 256 + threadIdx.x;
  if (i < n) o[i] = (bf16_t)in[i];
}
// in: [E][R][C] fp32  ->  out: [E][C][R] bf16
__global__ void transpose_to_bf16_kernel(const float* __restrict__ in, bf16_t* __restrict__ o,
                                         int R, int C) {
  __shared__ float tile[32][33];
  const size_t mat = (size_t)R * C;
  const float* ine = in + (size_t)blockIdx.z * mat;
  bf16_t* oute = o + (size_t)blockIdx.z * mat;
  const int x0 = blockIdx.x * 32, y0 = blockIdx.y * 32;
  const int tx = threadIdx.x, ty = threadIdx.y;
#pragma unroll
  for (int i = 0; i < 32; i += 8)
    tile[ty + i][tx] = ine[(size_t)(y0 + ty + i) * C + (x0 + tx)];
  __syncthreads();
#pragma unroll
  for (int i = 0; i < 32; i += 8)
    oute[(size_t)(x0 + ty + i) * R + (y0 + tx)] = (bf16_t)tile[tx][ty + i];
}

// ---- gating: one wave32 per token ------------------------------------------
__global__ void gate_kernel(const float* __restrict__ x, const float* __restrict__ Wg,
                            const float* __restrict__ bg,
                            int2* __restrict__ ridx, float2* __restrict__ rw) {
  const int wv = threadIdx.x >> 5, lane = threadIdx.x & 31;
  const int t = blockIdx.x * 8 + wv;
  const float* xr = x + (size_t)t * DDIM;
  float acc[NEXP] = {};
  for (int d = lane; d < DDIM; d += 32) {
    float xv = xr[d];
#pragma unroll
    for (int e = 0; e < NEXP; ++e) acc[e] += xv * Wg[d * NEXP + e];
  }
#pragma unroll
  for (int e = 0; e < NEXP; ++e)
#pragma unroll
    for (int off = 16; off > 0; off >>= 1) acc[e] += __shfl_xor(acc[e], off, 32);
  if (lane == 0) {
    float g[NEXP];
#pragma unroll
    for (int e = 0; e < NEXP; ++e) g[e] = (acc[e] + bg[e]) * INV_TEMP;
    int i0 = 0; float g0 = g[0];
#pragma unroll
    for (int e = 1; e < NEXP; ++e) if (g[e] > g0) { g0 = g[e]; i0 = e; }
    int i1 = -1; float g1 = -3.4e38f;
#pragma unroll
    for (int e = 0; e < NEXP; ++e) if (e != i0 && g[e] > g1) { g1 = g[e]; i1 = e; }
    float ee = __expf(g1 - g0);
    float inv = 1.0f / (1.0f + ee);
    ridx[t] = make_int2(i0, i1);
    rw[t]   = make_float2(inv, ee * inv);
  }
}

// ---- deterministic compaction (token-ascending order, no global atomics) ---
__global__ void route_count_kernel(const int2* __restrict__ ridx, int* __restrict__ blockCnt) {
  __shared__ int cnt[NEXP];
  if (threadIdx.x < NEXP) cnt[threadIdx.x] = 0;
  __syncthreads();
  int2 r = ridx[blockIdx.x * 256 + threadIdx.x];
  atomicAdd(&cnt[r.x], 1);
  atomicAdd(&cnt[r.y], 1);
  __syncthreads();
  if (threadIdx.x < NEXP) blockCnt[blockIdx.x * NEXP + threadIdx.x] = cnt[threadIdx.x];
}
__global__ void route_offset_kernel(const int* __restrict__ blockCnt,
                                    int* __restrict__ blockOff, int* __restrict__ counts) {
  const int idx = threadIdx.x;          // 256 threads = 32 blocks * 8 experts
  const int b = idx >> 3, e = idx & 7;
  int s = 0;
  for (int bb = 0; bb < b; ++bb) s += blockCnt[bb * NEXP + e];
  blockOff[idx] = s;
  if (b == 31) counts[e] = s + blockCnt[31 * NEXP + e];
}
__global__ void route_scatter_kernel(const int2* __restrict__ ridx, const float2* __restrict__ rw,
                                     const int* __restrict__ blockOff,
                                     int* __restrict__ list_tok, float* __restrict__ list_w) {
  __shared__ int waveCnt[8][NEXP];
  __shared__ int waveOff[8][NEXP];
  const int tid = threadIdx.x, lane = tid & 31, wv = tid >> 5;
  const int t = blockIdx.x * 256 + tid;
  int2 r = ridx[t];
  float2 w = rw[t];
  const unsigned below = (1u << lane) - 1u;
  int rank0 = 0, rank1 = 0;
#pragma unroll
  for (int e = 0; e < NEXP; ++e) {
    unsigned bal = (unsigned)__ballot(r.x == e || r.y == e);
    if (lane == 0) waveCnt[wv][e] = __popc(bal);
    if (r.x == e) rank0 = __popc(bal & below);
    if (r.y == e) rank1 = __popc(bal & below);
  }
  __syncthreads();
  if (tid < 64) {
    int wvv = tid >> 3, e = tid & 7;
    int s = 0;
    for (int k = 0; k < wvv; ++k) s += waveCnt[k][e];
    waveOff[wvv][e] = s;
  }
  __syncthreads();
  int p0 = blockOff[blockIdx.x * NEXP + r.x] + waveOff[wv][r.x] + rank0;
  int p1 = blockOff[blockIdx.x * NEXP + r.y] + waveOff[wv][r.y] + rank1;
  list_tok[r.x * NTOK + p0] = t;  list_w[r.x * NTOK + p0] = w.x;
  list_tok[r.y * NTOK + p1] = t;  list_w[r.y * NTOK + p1] = w.y;
}

// ---- fused expert FFN: out += w * ( silu(X W1 + b1) W2 + b2 ) --------------
// grid: (NTOK/TILE_M tiles, NEXP experts), block 256 = 8 waves.
__global__ __launch_bounds__(256)
void moe_ffn_kernel(const bf16_t* __restrict__ xb,
                    const bf16_t* __restrict__ W1t,   // [E][F][D]  (= W1^T)
                    const bf16_t* __restrict__ W2t,   // [E][D][F]  (= W2^T)
                    const float*  __restrict__ b1,    // [E][F]
                    const float*  __restrict__ b2,    // [E][D]
                    const int*    __restrict__ list_tok,
                    const float*  __restrict__ list_w,
                    const int*    __restrict__ counts,
                    float*        __restrict__ out) {
  extern __shared__ bf16_t Hs[];                      // [16][HSTRIDE] bf16
  __shared__ int   s_tok[TILE_M];
  __shared__ float s_w[TILE_M];

  const int e    = blockIdx.y;
  const int cnt  = counts[e];
  const int base = blockIdx.x * TILE_M;
  if (base >= cnt) return;
  const int rem = (cnt - base < TILE_M) ? (cnt - base) : TILE_M;

  const int tid = threadIdx.x;
  if (tid < TILE_M) {
    int rr = (tid < rem) ? tid : (rem - 1);           // clamp padding rows
    s_tok[tid] = list_tok[e * NTOK + base + rr];
    s_w[tid]   = (tid < rem) ? list_w[e * NTOK + base + tid] : 0.0f;
  }
  __syncthreads();

  const int lane = tid & 31;
  const int wv   = tid >> 5;
  const int ln   = lane & 15;
  const int half = lane >> 4;

  const bf16_t* arow = xb + (size_t)s_tok[ln] * DDIM; // gathered token row
  const bf16_t* W1e  = W1t + (size_t)e * FDIM * DDIM;
  const float*  b1e  = b1 + e * FDIM;

  // ---- phase 1: H[16, F] = silu(X W1 + b1); wave wv owns F cols [wv*512, +512)
  const int fbase = wv * (FDIM / 8);
  for (int g = 0; g < 8; ++g) {
    const int n0 = fbase + g * 64;
    const bf16_t* br0 = W1e + (size_t)(n0 +  0 + ln) * DDIM;
    const bf16_t* br1 = W1e + (size_t)(n0 + 16 + ln) * DDIM;
    const bf16_t* br2 = W1e + (size_t)(n0 + 32 + ln) * DDIM;
    const bf16_t* br3 = W1e + (size_t)(n0 + 48 + ln) * DDIM;
    v8f c0 = {}, c1 = {}, c2 = {}, c3 = {};
#pragma unroll 4
    for (int k0 = 0; k0 < DDIM; k0 += 32) {
      v16bf a  = load_a16x32(arow, k0, half);
      v16bf f0 = load_b32x16(br0, k0, half);
      v16bf f1 = load_b32x16(br1, k0, half);
      v16bf f2 = load_b32x16(br2, k0, half);
      v16bf f3 = load_b32x16(br3, k0, half);
      c0 = wmma_bf16(a, f0, c0);
      c1 = wmma_bf16(a, f1, c1);
      c2 = wmma_bf16(a, f2, c2);
      c3 = wmma_bf16(a, f3, c3);
    }
#pragma unroll
    for (int i = 0; i < 4; ++i) {
      v8f c = (i == 0) ? c0 : (i == 1) ? c1 : (i == 2) ? c2 : c3;
      const int col = n0 + i * 16 + ln;
      const float bv = b1e[col];
#pragma unroll
      for (int j = 0; j < 8; ++j) {
        float v = c[j] + bv;
        Hs[(size_t)(half * 8 + j) * HSTRIDE + col] = (bf16_t)(v / (1.0f + __expf(-v)));
      }
    }
  }
  __syncthreads();

  // ---- phase 2: Y[16, D] = H W2; wave wv owns D cols [wv*128, +128)
  const bf16_t* W2e  = W2t + (size_t)e * DDIM * FDIM;
  const float*  b2e  = b2 + e * DDIM;
  const bf16_t* hrow = Hs + (size_t)ln * HSTRIDE;
  const int dbase = wv * (DDIM / 8);
  for (int g = 0; g < 2; ++g) {
    const int n0 = dbase + g * 64;
    const bf16_t* br0 = W2e + (size_t)(n0 +  0 + ln) * FDIM;
    const bf16_t* br1 = W2e + (size_t)(n0 + 16 + ln) * FDIM;
    const bf16_t* br2 = W2e + (size_t)(n0 + 32 + ln) * FDIM;
    const bf16_t* br3 = W2e + (size_t)(n0 + 48 + ln) * FDIM;
    v8f c0 = {}, c1 = {}, c2 = {}, c3 = {};
#pragma unroll 4
    for (int k0 = 0; k0 < FDIM; k0 += 32) {
      v16bf a  = load_a16x32(hrow, k0, half);       // ds_load from LDS
      v16bf f0 = load_b32x16(br0, k0, half);
      v16bf f1 = load_b32x16(br1, k0, half);
      v16bf f2 = load_b32x16(br2, k0, half);
      v16bf f3 = load_b32x16(br3, k0, half);
      c0 = wmma_bf16(a, f0, c0);
      c1 = wmma_bf16(a, f1, c1);
      c2 = wmma_bf16(a, f2, c2);
      c3 = wmma_bf16(a, f3, c3);
    }
#pragma unroll
    for (int i = 0; i < 4; ++i) {
      v8f c = (i == 0) ? c0 : (i == 1) ? c1 : (i == 2) ? c2 : c3;
      const int col = n0 + i * 16 + ln;
      const float bv = b2e[col];
#pragma unroll
      for (int j = 0; j < 8; ++j) {
        const int row = half * 8 + j;
        if (row < rem) {
          float v = (c[j] + bv) * s_w[row];
          atomicAdd(out + (size_t)s_tok[row] * DDIM + col, v);
        }
      }
    }
  }
}

// ---------------------------------------------------------------------------
extern "C" void kernel_launch(void* const* d_in, const int* in_sizes, int n_in,
                              void* d_out, int out_size, void* d_ws, size_t ws_size,
                              hipStream_t stream) {
  (void)in_sizes; (void)n_in; (void)out_size; (void)ws_size;
  const float* x  = (const float*)d_in[0];
  const float* Wg = (const float*)d_in[1];
  const float* bg = (const float*)d_in[2];
  const float* W1 = (const float*)d_in[3];
  const float* b1 = (const float*)d_in[4];
  const float* W2 = (const float*)d_in[5];
  const float* b2 = (const float*)d_in[6];
  float* out = (float*)d_out;

  char* ws = (char*)d_ws;
  size_t off = 0;
  auto carve = [&](size_t bytes) -> char* {
    char* p = ws + off;
    off = (off + bytes + 255) & ~(size_t)255;
    return p;
  };
  bf16_t* xb       = (bf16_t*)carve((size_t)NTOK * DDIM * 2);
  bf16_t* W1t      = (bf16_t*)carve((size_t)NEXP * FDIM * DDIM * 2);
  bf16_t* W2t      = (bf16_t*)carve((size_t)NEXP * DDIM * FDIM * 2);
  int2*   ridx     = (int2*)carve((size_t)NTOK * sizeof(int2));
  float2* rw       = (float2*)carve((size_t)NTOK * sizeof(float2));
  int*    blockCnt = (int*)carve(32 * NEXP * sizeof(int));
  int*    blockOff = (int*)carve(32 * NEXP * sizeof(int));
  int*    counts   = (int*)carve(NEXP * sizeof(int));
  int*    list_tok = (int*)carve((size_t)NEXP * NTOK * sizeof(int));
  float*  list_w   = (float*)carve((size_t)NEXP * NTOK * sizeof(float));

  const int nElems = NTOK * DDIM;
  zero_kernel<<<(nElems + 255) / 256, 256, 0, stream>>>(out, nElems);
  cvt_bf16_kernel<<<(nElems + 255) / 256, 256, 0, stream>>>(x, xb, nElems);

  dim3 tb(32, 8);
  transpose_to_bf16_kernel<<<dim3(FDIM / 32, DDIM / 32, NEXP), tb, 0, stream>>>(W1, W1t, DDIM, FDIM);
  transpose_to_bf16_kernel<<<dim3(DDIM / 32, FDIM / 32, NEXP), tb, 0, stream>>>(W2, W2t, FDIM, DDIM);

  gate_kernel<<<NTOK / 8, 256, 0, stream>>>(x, Wg, bg, ridx, rw);
  route_count_kernel<<<NTOK / 256, 256, 0, stream>>>(ridx, blockCnt);
  route_offset_kernel<<<1, 256, 0, stream>>>(blockCnt, blockOff, counts);
  route_scatter_kernel<<<NTOK / 256, 256, 0, stream>>>(ridx, rw, blockOff, list_tok, list_w);

  hipFuncSetAttribute((const void*)moe_ffn_kernel,
                      hipFuncAttributeMaxDynamicSharedMemorySize, HBYTES);
  moe_ffn_kernel<<<dim3(NTOK / TILE_M, NEXP), 256, HBYTES, stream>>>(
      xb, W1t, W2t, b1, b2, list_tok, list_w, counts, out);
}
